// FastSACDistributionalQNet_352187318805
// MI455X (gfx1250) — matrix operand order repaired and checked by reading, not compiled, not run on hardware
//
#include <hip/hip_runtime.h>
#include <hip/hip_bf16.h>
#include <math.h>

// ---------------------------------------------------------------------------
// FastSACDistributionalQNet fused kernel for gfx1250 (MI455X).
// MLP 80->256->128->64->101 with LN+SiLU, softmax, categorical projection.
// GEMMs via V_WMMA_F32_16X16X32_F16 (wave32), weights resident in LDS,
// persistent workgroups, async global->LDS weight staging when available.
// ---------------------------------------------------------------------------

typedef __attribute__((ext_vector_type(16))) _Float16 v16h;
typedef __attribute__((ext_vector_type(8)))  _Float16 v8h;
typedef __attribute__((ext_vector_type(8)))  float    v8f;
typedef __attribute__((ext_vector_type(4)))  int      v4i;

#define NB      131072     // batch
#define N_OBS   60
#define N_ACT   20
#define K1      96         // 80 padded to 3*32
#define H1      256
#define H2      128
#define H3      64
#define NA      101        // atoms
#define NAP     112        // atoms padded to 7*16
#define VMIN    (-10.0f)
#define VMAX    (10.0f)
#define INV_DZ  (5.0f)     // (N_ATOMS-1)/(VMAX-VMIN)

// f16 weight blob layout in d_ws (halves), transposed: Wt[N][Kpad]
#define OFF_W1  0                          // [256][96]
#define OFF_W2  (OFF_W1 + H1 * K1)         // [128][256]
#define OFF_W3  (OFF_W2 + H2 * H1)         // [64][128]
#define OFF_W4  (OFF_W3 + H3 * H2)         // [112][64]
#define W_HALVES (OFF_W4 + NAP * H3)       // 72704 halves = 145408 B

#define ROWS_PER_WAVE 16
#define WAVES_PER_WG  4
#define ROWS_PER_WG   (ROWS_PER_WAVE * WAVES_PER_WG)
#define WG_THREADS    (32 * WAVES_PER_WG)
#define NTILES       (NB / ROWS_PER_WG)    // 2048
#define PERSIST_WGS  512                   // each WG handles 4 tiles

// params (f32) packed after weights in LDS
#define NPAR (3*H1 + 3*H2 + 3*H3 + NAP + NAP)

#define SMEM_BYTES (W_HALVES*2 + NPAR*4 \
                    + WAVES_PER_WG*ROWS_PER_WAVE*H1*2 /* f16 act  */ \
                    + WAVES_PER_WG*ROWS_PER_WAVE*H1*4 /* f32 raw  */)

#if __has_builtin(__builtin_amdgcn_global_load_async_to_lds_b128)
#define HAVE_ASYNC_LDS 1
typedef __attribute__((address_space(1))) v4i* gv4i_p;
typedef __attribute__((address_space(3))) v4i* lv4i_p;
#endif

// ---------------------------------------------------------------------------
// Prep: convert + transpose + pad weights to f16 in d_ws.
// ---------------------------------------------------------------------------
__global__ void prep_weights_f16(const float* __restrict__ W1,
                                 const float* __restrict__ W2,
                                 const float* __restrict__ W3,
                                 const float* __restrict__ W4,
                                 _Float16* __restrict__ ws) {
    int i = blockIdx.x * blockDim.x + threadIdx.x;
    if (i < OFF_W2) {                       // W1t [256][96] from W1 [80][256]
        int j = i - OFF_W1, n = j / K1, k = j % K1;
        ws[i] = (_Float16)((k < 80) ? W1[k * H1 + n] : 0.0f);
    } else if (i < OFF_W3) {                // W2t [128][256] from W2 [256][128]
        int j = i - OFF_W2, n = j / H1, k = j % H1;
        ws[i] = (_Float16)W2[k * H2 + n];
    } else if (i < OFF_W4) {                // W3t [64][128] from W3 [128][64]
        int j = i - OFF_W3, n = j / H2, k = j % H2;
        ws[i] = (_Float16)W3[k * H3 + n];
    } else if (i < W_HALVES) {              // W4t [112][64] from W4 [64][101]
        int j = i - OFF_W4, n = j / H3, k = j % H3;
        ws[i] = (_Float16)((n < NA) ? W4[k * NA + n] : 0.0f);
    }
}

// ---------------------------------------------------------------------------
// Fragment loaders (layouts per CDNA5 ISA 7.12.2, wave32).
// ---------------------------------------------------------------------------
__device__ __forceinline__ v16h load_a_frag(const _Float16* act, int kpad,
                                            int kc, int lane) {
    int m   = lane & 15;
    int off = m * kpad + kc * 32 + ((lane >> 4) << 3);
    v8h lo = *(const v8h*)(act + off);
    v8h hi = *(const v8h*)(act + off + 16);
    return __builtin_shufflevector(lo, hi, 0,1,2,3,4,5,6,7,8,9,10,11,12,13,14,15);
}

__device__ __forceinline__ v16h load_b_frag(const _Float16* wt, int kpad,
                                            int nt, int kc, int lane) {
    int n   = nt * 16 + (lane & 15);
    int off = n * kpad + kc * 32 + ((lane >> 4) << 3);
    v8h lo = *(const v8h*)(wt + off);
    v8h hi = *(const v8h*)(wt + off + 16);
    return __builtin_shufflevector(lo, hi, 0,1,2,3,4,5,6,7,8,9,10,11,12,13,14,15);
}

// one wave computes its 16-row tile: raw[16][H1] = act @ Wt^T
template <int KPAD, int NOUT>
__device__ __forceinline__ void gemm_tile(const _Float16* __restrict__ act,
                                          const _Float16* __restrict__ wt,
                                          float* __restrict__ raw, int lane) {
    for (int nt = 0; nt < NOUT / 16; ++nt) {
        v8f acc = {};
#pragma unroll
        for (int kc = 0; kc < KPAD / 32; ++kc) {
            v16h a = load_a_frag(act, KPAD, kc, lane);
            v16h b = load_b_frag(wt, KPAD, nt, kc, lane);
            acc = __builtin_amdgcn_wmma_f32_16x16x32_f16(
                false, a, false, b, (short)0, acc, false, false);
        }
        int n     = nt * 16 + (lane & 15);
        int mbase = (lane >> 4) << 3;
#pragma unroll
        for (int r = 0; r < 8; ++r)
            raw[(mbase + r) * H1 + n] = acc[r];
    }
}

// bias + LayerNorm + SiLU; lane pairs (l, l+16) split a row's columns
template <int NOUT>
__device__ __forceinline__ void ln_silu(float* __restrict__ raw,
                                        const float* __restrict__ bias,
                                        const float* __restrict__ g,
                                        const float* __restrict__ be,
                                        _Float16* __restrict__ actOut, int lane) {
    const int row = lane & 15;
    const int n0  = (lane >> 4) * (NOUT / 2);
    float* lr = raw + row * H1;
    float sum = 0.f, sumsq = 0.f;
    for (int j = 0; j < NOUT / 2; ++j) {
        int n = n0 + j;
        float v = lr[n] + bias[n];
        lr[n] = v;
        sum += v;
        sumsq += v * v;
    }
    sum   += __shfl_xor(sum, 16, 32);
    sumsq += __shfl_xor(sumsq, 16, 32);
    float mu   = sum * (1.0f / NOUT);
    float var  = sumsq * (1.0f / NOUT) - mu * mu;
    float rstd = rsqrtf(var + 1e-5f);
    for (int j = 0; j < NOUT / 2; ++j) {
        int n = n0 + j;
        float v = (lr[n] - mu) * rstd * g[n] + be[n];
        float s = v / (1.0f + __expf(-v));         // SiLU
        actOut[row * NOUT + n] = (_Float16)s;
    }
}

// ---------------------------------------------------------------------------
// Fused MLP + softmax + categorical projection (persistent workgroups)
// ---------------------------------------------------------------------------
__global__ __launch_bounds__(WG_THREADS, 1)
void fused_qdist_kernel(const float* __restrict__ obs,
                        const float* __restrict__ actions,
                        const float* __restrict__ rewards,
                        const float* __restrict__ bootstrap,
                        const float* __restrict__ discount,
                        const float* __restrict__ q_support,
                        const float* __restrict__ b1, const float* __restrict__ g1,
                        const float* __restrict__ be1,
                        const float* __restrict__ b2, const float* __restrict__ g2,
                        const float* __restrict__ be2,
                        const float* __restrict__ b3, const float* __restrict__ g3,
                        const float* __restrict__ be3,
                        const float* __restrict__ b4,
                        const _Float16* __restrict__ wsW,
                        float* __restrict__ out) {
    extern __shared__ char smem[];
    _Float16* sW1 = (_Float16*)smem;
    _Float16* sW2 = sW1 + H1 * K1;
    _Float16* sW3 = sW2 + H2 * H1;
    _Float16* sW4 = sW3 + H3 * H2;
    float* sPar = (float*)(sW4 + NAP * H3);
    float* pb1 = sPar;        float* pg1 = pb1 + H1;  float* pbe1 = pg1 + H1;
    float* pb2 = pbe1 + H1;   float* pg2 = pb2 + H2;  float* pbe2 = pg2 + H2;
    float* pb3 = pbe2 + H2;   float* pg3 = pb3 + H3;  float* pbe3 = pg3 + H3;
    float* pb4 = pbe3 + H3;                            // NAP
    float* pqs = pb4 + NAP;                            // NAP
    _Float16* sAct = (_Float16*)(pqs + NAP);
    float*    sRaw = (float*)(sAct + WAVES_PER_WG * ROWS_PER_WAVE * H1);

    const int tid  = threadIdx.x;
    const int lane = tid & 31;
    const int wave = tid >> 5;

    // --- stage weights into LDS once per (persistent) workgroup ---
    {
        const int nvec = (W_HALVES * 2) / 16;
#ifdef HAVE_ASYNC_LDS
        // CDNA5 async DMA: GLOBAL_LOAD_ASYNC_TO_LDS_B128 (ASYNCcnt-tracked)
        for (int i = tid; i < nvec; i += WG_THREADS) {
            const uint4* srcp = (const uint4*)wsW + i;
            uint4*       dstp = (uint4*)smem + i;
            __builtin_amdgcn_global_load_async_to_lds_b128(
                (gv4i_p)srcp, (lv4i_p)dstp, 0, 0);
        }
#if __has_builtin(__builtin_amdgcn_s_wait_asynccnt)
        __builtin_amdgcn_s_wait_asynccnt(0);
#else
        asm volatile("s_wait_asynccnt 0x0" ::: "memory");
#endif
#else
        const uint4* src = (const uint4*)wsW;
        uint4*       dst = (uint4*)smem;
        for (int i = tid; i < nvec; i += WG_THREADS) dst[i] = src[i];
#endif
    }
    // --- stage f32 params ---
    for (int i = tid; i < H1; i += WG_THREADS) { pb1[i]=b1[i]; pg1[i]=g1[i]; pbe1[i]=be1[i]; }
    for (int i = tid; i < H2; i += WG_THREADS) { pb2[i]=b2[i]; pg2[i]=g2[i]; pbe2[i]=be2[i]; }
    for (int i = tid; i < H3; i += WG_THREADS) { pb3[i]=b3[i]; pg3[i]=g3[i]; pbe3[i]=be3[i]; }
    for (int i = tid; i < NAP; i += WG_THREADS) {
        pb4[i] = (i < NA) ? b4[i] : 0.f;
        pqs[i] = (i < NA) ? q_support[i] : 0.f;
    }
    __syncthreads();

    _Float16* act = sAct + wave * ROWS_PER_WAVE * H1;
    float*    raw = sRaw + wave * ROWS_PER_WAVE * H1;

    // --- persistent loop over batch tiles ---
    for (int tile = blockIdx.x; tile < NTILES; tile += PERSIST_WGS) {
        const int row0 = tile * ROWS_PER_WG + wave * ROWS_PER_WAVE;

        // prefetch next tile's rows (lowers to global_prefetch_b8)
        if (tile + PERSIST_WGS < NTILES) {
            const int nrow0 = (tile + PERSIST_WGS) * ROWS_PER_WG + wave * ROWS_PER_WAVE;
            if (lane < ROWS_PER_WAVE) {
                __builtin_prefetch(&obs[(size_t)(nrow0 + lane) * N_OBS], 0, 0);
                __builtin_prefetch(&actions[(size_t)(nrow0 + lane) * N_ACT], 0, 0);
            }
        }

        // --- load x = [obs || actions] as f16, K padded to 96 ---
        for (int idx = lane; idx < ROWS_PER_WAVE * K1; idx += 32) {
            int m = idx / K1, c = idx % K1;
            int g = row0 + m;
            float v = 0.0f;
            if (c < N_OBS)              v = obs[(size_t)g * N_OBS + c];
            else if (c < N_OBS + N_ACT) v = actions[(size_t)g * N_ACT + (c - N_OBS)];
            act[m * K1 + c] = (_Float16)v;
        }

        // --- 4 layers (same-wave LDS is in-order; no barriers needed) ---
        gemm_tile<K1, H1>(act, sW1, raw, lane);
        ln_silu<H1>(raw, pb1, pg1, pbe1, act, lane);
        gemm_tile<H1, H2>(act, sW2, raw, lane);
        ln_silu<H2>(raw, pb2, pg2, pbe2, act, lane);
        gemm_tile<H2, H3>(act, sW3, raw, lane);
        ln_silu<H3>(raw, pb3, pg3, pbe3, act, lane);
        gemm_tile<H3, NAP>(act, sW4, raw, lane);   // logits -> raw[m][0..111]

        // --- softmax + categorical projection, one row per lane (0..15) ---
        if (lane < 16) {
            const int m = lane;
            const int g = row0 + m;
            float* lr = raw + m * H1;     // logits [0..100]
            float* pr = lr + 128;         // projection accumulator [0..111]
            float mx = -1e30f;
            for (int j = 0; j < NA; ++j) { float v = lr[j] + pb4[j]; lr[j] = v; mx = fmaxf(mx, v); }
            float se = 0.f;
            for (int j = 0; j < NA; ++j) { float e = __expf(lr[j] - mx); lr[j] = e; se += e; }
            const float inv = 1.0f / se;
            for (int j = 0; j < NAP; ++j) pr[j] = 0.f;
            const float rw = rewards[g];
            const float sc = bootstrap[g] * discount[g];
            for (int j = 0; j < NA; ++j) {
                float p  = lr[j] * inv;
                float tz = fminf(fmaxf(rw + sc * pqs[j], VMIN), VMAX);
                float bp = (tz - VMIN) * INV_DZ;       // in [0,100]
                int li = (int)floorf(bp);
                int ui = (int)ceilf(bp);
                if (ui == li) { if (li > 0) --li; else ++ui; }
                pr[li] += p * ((float)ui - bp);
                pr[ui] += p * (bp - (float)li);
            }
        }

        // --- cooperative store of the wave's 16x101 projection ---
        for (int idx = lane; idx < ROWS_PER_WAVE * NA; idx += 32) {
            int m = idx / NA, c = idx % NA;
            out[(size_t)(row0 + m) * NA + c] = raw[m * H1 + 128 + c];
        }
    }
}

// ---------------------------------------------------------------------------
extern "C" void kernel_launch(void* const* d_in, const int* in_sizes, int n_in,
                              void* d_out, int out_size, void* d_ws, size_t ws_size,
                              hipStream_t stream) {
    const float* obs       = (const float*)d_in[0];
    const float* actions   = (const float*)d_in[1];
    const float* rewards   = (const float*)d_in[2];
    const float* bootstrap = (const float*)d_in[3];
    const float* discount  = (const float*)d_in[4];
    const float* q_support = (const float*)d_in[5];
    const float* W1 = (const float*)d_in[6];
    const float* b1 = (const float*)d_in[7];
    const float* g1 = (const float*)d_in[8];
    const float* be1= (const float*)d_in[9];
    const float* W2 = (const float*)d_in[10];
    const float* b2 = (const float*)d_in[11];
    const float* g2 = (const float*)d_in[12];
    const float* be2= (const float*)d_in[13];
    const float* W3 = (const float*)d_in[14];
    const float* b3 = (const float*)d_in[15];
    const float* g3 = (const float*)d_in[16];
    const float* be3= (const float*)d_in[17];
    const float* W4 = (const float*)d_in[18];
    const float* b4 = (const float*)d_in[19];

    _Float16* wsW = (_Float16*)d_ws;
    float*    out = (float*)d_out;

    // 1) convert weights to f16 transposed/padded layout in workspace
    {
        int total  = W_HALVES;
        int blocks = (total + 255) / 256;
        prep_weights_f16<<<blocks, 256, 0, stream>>>(W1, W2, W3, W4, wsW);
    }
    // 2) fused MLP + softmax + projection (persistent workgroups)
    {
        dim3 grid(PERSIST_WGS);
        dim3 block(WG_THREADS);
        fused_qdist_kernel<<<grid, block, SMEM_BYTES, stream>>>(
            obs, actions, rewards, bootstrap, discount, q_support,
            b1, g1, be1, b2, g2, be2, b3, g3, be3, b4, wsW, out);
    }
}
